// Attention_86088324481794
// MI455X (gfx1250) — matrix operand order repaired and checked by reading, not compiled, not run on hardware
//
#include <hip/hip_runtime.h>

// Problem constants (match reference)
#define BB 4
#define NN 1024
#define DIMK 1024
#define HH 16
#define HDM 64
#define BN (BB * NN)
#define DCHUNKS 64

typedef unsigned short u16;
typedef __bf16 bf16;
typedef __attribute__((ext_vector_type(16))) __bf16 v16bf;
typedef __attribute__((ext_vector_type(8)))  __bf16 v8bf;
typedef __attribute__((ext_vector_type(8)))  float  v8f;

__device__ __forceinline__ u16 f2bfu(float f) {
  unsigned u = __builtin_bit_cast(unsigned, f);
  unsigned r = (u + 0x7FFFu + ((u >> 16) & 1u)) >> 16;  // RNE
  return (u16)r;
}
__device__ __forceinline__ bf16 f2bf(float f) {
  u16 r = f2bfu(f);
  return __builtin_bit_cast(bf16, r);
}

#define WMMA_BF16(a, b, c) \
  __builtin_amdgcn_wmma_f32_16x16x32_bf16(false, (a), false, (b), (short)0, (c), false, false)

// ---------------------------------------------------------------------------
// Small prep kernels
// ---------------------------------------------------------------------------

// jax bool arrays are 1 byte/element
__global__ void kmf_kernel(const unsigned char* __restrict__ mask,
                           float* __restrict__ kmf, int n) {
  int i = blockIdx.x * blockDim.x + threadIdx.x;
  if (i < n) kmf[i] = mask[i] ? 1.0f : 0.0f;
}

__global__ void f32_to_bf16_kernel(const float* __restrict__ in,
                                   u16* __restrict__ out, int n) {
  int i = blockIdx.x * blockDim.x + threadIdx.x;
  if (i < n) out[i] = f2bfu(in[i]);
}

// wT[n*K + k] = bf16(w[k*Nc + n])   (store weights transposed for B-fragments)
__global__ void wtrans_kernel(const float* __restrict__ w, u16* __restrict__ wT,
                              int K, int Nc) {
  int i = blockIdx.x * blockDim.x + threadIdx.x;
  if (i < K * Nc) {
    int n = i / K, k = i % K;
    wT[i] = f2bfu(w[(size_t)k * Nc + n]);
  }
}

// Kswap[2t] = K[2t+1] ; Kswap[2t+1] = -K[2t]   (sign flip on bf16 bits)
__global__ void kswap_kernel(const u16* __restrict__ Kb, u16* __restrict__ Ks,
                             int npairs) {
  int p = blockIdx.x * blockDim.x + threadIdx.x;
  if (p < npairs) {
    u16 a = Kb[2 * p], b = Kb[2 * p + 1];
    Ks[2 * p] = b;
    Ks[2 * p + 1] = (u16)(a ^ 0x8000u);
  }
}

// Vt[((b*H+h)*HD+d)*N + t] = V[((b*N+t)*H+h)*HD + d]
__global__ void vtrans_kernel(const u16* __restrict__ V, u16* __restrict__ Vt,
                              int total) {
  int i = blockIdx.x * blockDim.x + threadIdx.x;
  if (i < total) {
    int t = i % NN;
    int d = (i / NN) % HDM;
    int h = (i / (NN * HDM)) % HH;
    int b = i / (NN * HDM * HH);
    Vt[i] = V[(((size_t)b * NN + t) * HH + h) * HDM + d];
  }
}

// partial[b*DCHUNKS + chunk] = sum over slice of d_ij * km_i * km_j
__global__ void dist_partial_kernel(const float* __restrict__ dist,
                                    const float* __restrict__ kmf,
                                    float* __restrict__ partial) {
  int b = blockIdx.y;
  int chunk = blockIdx.x;
  const int per = (NN * NN) / DCHUNKS;  // 16384
  const float* D = dist + (size_t)b * NN * NN;
  const float* Km = kmf + (size_t)b * NN;
  float acc = 0.f;
  for (int t = threadIdx.x; t < per; t += blockDim.x) {
    int g = chunk * per + t;
    int i = g >> 10, j = g & (NN - 1);
    acc += D[g] * Km[i] * Km[j];
  }
  __shared__ float red[256];
  red[threadIdx.x] = acc;
  __syncthreads();
  for (int s = 128; s > 0; s >>= 1) {
    if (threadIdx.x < s) red[threadIdx.x] += red[threadIdx.x + s];
    __syncthreads();
  }
  if (threadIdx.x == 0) partial[b * DCHUNKS + chunk] = red[0];
}

// stats[0..B-1] = 1/mean_b ; stats[B] = softplus(gate_alpha)
__global__ void finalize_kernel(const float* __restrict__ partial,
                                const float* __restrict__ kmf,
                                const float* __restrict__ galpha,
                                float* __restrict__ stats) {
  __shared__ float red[256];
  for (int b = 0; b < BB; ++b) {
    float a = 0.f;
    for (int i = threadIdx.x; i < NN; i += 256) a += kmf[b * NN + i];
    red[threadIdx.x] = a;
    __syncthreads();
    for (int s = 128; s > 0; s >>= 1) {
      if (threadIdx.x < s) red[threadIdx.x] += red[threadIdx.x + s];
      __syncthreads();
    }
    float scount = red[0];
    __syncthreads();
    float p = 0.f;
    for (int i = threadIdx.x; i < DCHUNKS; i += 256) p += partial[b * DCHUNKS + i];
    red[threadIdx.x] = p;
    __syncthreads();
    for (int s = 128; s > 0; s >>= 1) {
      if (threadIdx.x < s) red[threadIdx.x] += red[threadIdx.x + s];
      __syncthreads();
    }
    if (threadIdx.x == 0) {
      float denom = fmaxf(scount * scount, 1.0f);
      float mean = fmaxf(red[0] / denom, 1e-6f);
      stats[b] = 1.0f / mean;
    }
    __syncthreads();
  }
  if (threadIdx.x == 0) stats[BB] = log1pf(__expf(galpha[0]));
}

// ---------------------------------------------------------------------------
// WMMA bf16 GEMM: C(MxNc,f32 accum) = A(MxK bf16, row-major) * W(KxNc) with
// W supplied transposed (WT: Nc x K row-major). 1 wave -> 16x64 output tile.
// ---------------------------------------------------------------------------
template <bool OUT_BF16>
__global__ __launch_bounds__(32) void gemm_bf16_kernel(
    const u16* __restrict__ A, const u16* __restrict__ WT,
    void* __restrict__ Cout, int M, int Nc, int K) {
  const int lane = threadIdx.x & 31;
  const int half = lane >> 4;
  const int l15 = lane & 15;
  const int rbase = half * 8;
  const int M0 = blockIdx.x * 16;
  const int n0 = blockIdx.y * 64;

  v8f vz = {0.f, 0.f, 0.f, 0.f, 0.f, 0.f, 0.f, 0.f};
  v8f acc[4];
  acc[0] = vz; acc[1] = vz; acc[2] = vz; acc[3] = vz;

  const u16* arow = A + (size_t)(M0 + l15) * K;
  const int o0 = half ? 16 : 0;
  const int o1 = half ? 24 : 8;
  const int bo = half ? 16 : 0;

  for (int kc = 0; kc < K; kc += 32) {
    union { v16bf v; v8bf h2[2]; } ua;
    ua.h2[0] = *(const v8bf*)(arow + kc + o0);
    ua.h2[1] = *(const v8bf*)(arow + kc + o1);
    const u16* wbase = WT + kc + bo;
#pragma unroll
    for (int c = 0; c < 4; ++c) {
      v16bf bfr = *(const v16bf*)(wbase + (size_t)(n0 + c * 16 + l15) * K);
      acc[c] = WMMA_BF16(ua.v, bfr, acc[c]);
    }
  }
#pragma unroll
  for (int c = 0; c < 4; ++c) {
#pragma unroll
    for (int r = 0; r < 8; ++r) {
      size_t idx = (size_t)(M0 + rbase + r) * Nc + n0 + c * 16 + l15;
      float v = acc[c][r];
      if (OUT_BF16) ((u16*)Cout)[idx] = f2bfu(v);
      else          ((float*)Cout)[idx] = v;
    }
  }
}

// ---------------------------------------------------------------------------
// Fused distance-RoPE attention, flash style. 1 wave per (b,h,16-query tile).
// Tracks running max m, Z=sum(exp), G=sum(exp*gate); acc = sum(exp*gate*V).
// out_row = acc / (G + 1e-6*Z), then * km_i, stored as bf16.
// ---------------------------------------------------------------------------
__global__ __launch_bounds__(32) void attn_kernel(
    const u16* __restrict__ Qb, const u16* __restrict__ Kb,
    const u16* __restrict__ Ksw, const u16* __restrict__ Vt,
    const float* __restrict__ dist, const float* __restrict__ kmf,
    const float* __restrict__ stats, const float* __restrict__ head_omega,
    u16* __restrict__ Ob) {
  const int lane = threadIdx.x & 31;
  const int half = lane >> 4;
  const int l15 = lane & 15;
  const int rbase = half * 8;
  const int it = blockIdx.x, h = blockIdx.y, b = blockIdx.z;
  const int i0 = it * 16;

  const float invmean = stats[b];
  const float alpha = stats[BB];
  const float om = head_omega[h] * invmean;  // theta = d_raw * om
  const float ai = alpha * invmean;          // gate  = exp(-ai * d_raw)

  const int ldt = HH * HDM;
  const u16* Qp = Qb + ((size_t)(b * NN + i0) * ldt + h * HDM);
  const u16* Kp = Kb + ((size_t)(b * NN) * ldt + h * HDM);
  const u16* Sp = Ksw + ((size_t)(b * NN) * ldt + h * HDM);
  const u16* Vp = Vt + ((size_t)(b * HH + h) * HDM) * NN;
  const float* Dp = dist + (size_t)b * NN * NN;
  const float* Km = kmf + (size_t)b * NN;

  const int o0 = half ? 16 : 0;
  const int o1 = half ? 24 : 8;

  // Q A-fragments (d 0..31 and d 32..63)
  v16bf aq0, aq1;
  {
    const u16* row = Qp + (size_t)l15 * ldt;
    union { v16bf v; v8bf h2[2]; } u;
    u.h2[0] = *(const v8bf*)(row + o0);
    u.h2[1] = *(const v8bf*)(row + o1);
    aq0 = u.v;
    u.h2[0] = *(const v8bf*)(row + 32 + o0);
    u.h2[1] = *(const v8bf*)(row + 32 + o1);
    aq1 = u.v;
  }

  float mrow[8], Zr[8], Gr[8];
  v8f vz = {0.f, 0.f, 0.f, 0.f, 0.f, 0.f, 0.f, 0.f};
  v8f acc[4];
  acc[0] = vz; acc[1] = vz; acc[2] = vz; acc[3] = vz;
#pragma unroll
  for (int r = 0; r < 8; ++r) { mrow[r] = -1e30f; Zr[r] = 0.f; Gr[r] = 0.f; }

  __shared__ float Plds[16][33];

  for (int kb = 0; kb < NN / 32; ++kb) {
    float sv[2][8], gv[2][8];
#pragma unroll
    for (int s = 0; s < 2; ++s) {
      const int j0 = kb * 32 + s * 16;
      const int j = j0 + l15;
      const u16* krow = Kp + (size_t)(j0 + l15) * ldt + (half ? 16 : 0);
      const u16* srow = Sp + (size_t)(j0 + l15) * ldt + (half ? 16 : 0);
      v16bf bk0 = *(const v16bf*)(krow);
      v16bf bk1 = *(const v16bf*)(krow + 32);
      v16bf bs0 = *(const v16bf*)(srow);
      v16bf bs1 = *(const v16bf*)(srow + 32);

      v8f cc = vz, cs = vz;
      cc = WMMA_BF16(aq0, bk0, cc);
      cc = WMMA_BF16(aq1, bk1, cc);
      cs = WMMA_BF16(aq0, bs0, cs);
      cs = WMMA_BF16(aq1, bs1, cs);

      const float kmj = Km[j];
#pragma unroll
      for (int r = 0; r < 8; ++r) {
        const int row = i0 + rbase + r;
        float d = Dp[(size_t)row * NN + j];
        float th = d * om;
        float sc = (cc[r] * __cosf(th) + cs[r] * __sinf(th)) * 0.125f;
        float g = __expf(-ai * d) * kmj;
        if (row == j) g = 1.0f;  // self-connection forced to 1
        sv[s][r] = (kmj > 0.f) ? sc : -1e30f;
        gv[s][r] = g;
      }
    }

    // online softmax update (row stats live across the 16 lanes of each half)
    float scale[8];
#pragma unroll
    for (int r = 0; r < 8; ++r) {
      float t = fmaxf(sv[0][r], sv[1][r]);
      t = fmaxf(t, __shfl_xor(t, 1, 16));
      t = fmaxf(t, __shfl_xor(t, 2, 16));
      t = fmaxf(t, __shfl_xor(t, 4, 16));
      t = fmaxf(t, __shfl_xor(t, 8, 16));
      float mnew = fmaxf(mrow[r], t);
      scale[r] = __expf(mrow[r] - mnew);
      mrow[r] = mnew;
      float e0 = (sv[0][r] > -1e29f) ? __expf(sv[0][r] - mnew) : 0.f;
      float e1 = (sv[1][r] > -1e29f) ? __expf(sv[1][r] - mnew) : 0.f;
      float p0 = e0 * gv[0][r], p1 = e1 * gv[1][r];
      float se = e0 + e1, sp = p0 + p1;
      se += __shfl_xor(se, 1, 16); se += __shfl_xor(se, 2, 16);
      se += __shfl_xor(se, 4, 16); se += __shfl_xor(se, 8, 16);
      sp += __shfl_xor(sp, 1, 16); sp += __shfl_xor(sp, 2, 16);
      sp += __shfl_xor(sp, 4, 16); sp += __shfl_xor(sp, 8, 16);
      Zr[r] = Zr[r] * scale[r] + se;
      Gr[r] = Gr[r] * scale[r] + sp;
      Plds[rbase + r][l15] = p0;
      Plds[rbase + r][16 + l15] = p1;
    }
#pragma unroll
    for (int c = 0; c < 4; ++c)
#pragma unroll
      for (int r = 0; r < 8; ++r) acc[c][r] *= scale[r];

    __syncthreads();
    // P A-fragment (16x32 keys) from LDS
    union { v16bf v; bf16 e[16]; } up;
#pragma unroll
    for (int t = 0; t < 8; ++t) {
      up.e[t] = f2bf(Plds[l15][o0 + t]);
      up.e[8 + t] = f2bf(Plds[l15][o1 + t]);
    }
    __syncthreads();

#pragma unroll
    for (int c = 0; c < 4; ++c) {
      const u16* vp = Vp + (size_t)(c * 16 + l15) * NN + kb * 32 + (half ? 16 : 0);
      acc[c] = WMMA_BF16(up.v, *(const v16bf*)vp, acc[c]);
    }
  }

  // epilogue: normalize, apply query mask, store bf16
#pragma unroll
  for (int r = 0; r < 8; ++r) {
    const int row = i0 + rbase + r;
    float denom = Gr[r] + 1e-6f * Zr[r];
    float kmi = Km[row];
    float invd = (denom > 0.f) ? (kmi / denom) : 0.f;
#pragma unroll
    for (int c = 0; c < 4; ++c) {
      float v = acc[c][r] * invd;
      Ob[((size_t)(b * NN + row) * HH + h) * HDM + c * 16 + l15] = f2bfu(v);
    }
  }
}

// ---------------------------------------------------------------------------
// Host launcher
// ---------------------------------------------------------------------------
extern "C" void kernel_launch(void* const* d_in, const int* in_sizes, int n_in,
                              void* d_out, int out_size, void* d_ws, size_t ws_size,
                              hipStream_t stream) {
  (void)in_sizes; (void)n_in; (void)out_size; (void)ws_size;
  const float* x = (const float*)d_in[0];
  const float* dist = (const float*)d_in[1];
  const unsigned char* mask = (const unsigned char*)d_in[2];  // jax bool = 1B
  const float* wq = (const float*)d_in[3];
  const float* wk = (const float*)d_in[4];
  const float* wv = (const float*)d_in[5];
  const float* wo = (const float*)d_in[6];
  const float* head_omega = (const float*)d_in[7];
  const float* galpha = (const float*)d_in[8];
  float* out = (float*)d_out;

  char* ws = (char*)d_ws;
  size_t off = 0;
  auto alloc = [&](size_t bytes) -> void* {
    off = (off + 255) & ~(size_t)255;
    void* p = ws + off;
    off += bytes;
    return p;
  };

  u16* xb  = (u16*)alloc((size_t)BN * DIMK * 2);   // bf16 x; reused as attn-out
  u16* wqT = (u16*)alloc((size_t)DIMK * DIMK * 2);
  u16* wkT = (u16*)alloc((size_t)DIMK * DIMK * 2);
  u16* wvT = (u16*)alloc((size_t)DIMK * DIMK * 2);
  u16* woT = (u16*)alloc((size_t)DIMK * DIMK * 2);
  u16* Qb  = (u16*)alloc((size_t)BN * DIMK * 2);
  u16* Kb  = (u16*)alloc((size_t)BN * DIMK * 2);
  u16* Ksw = (u16*)alloc((size_t)BN * DIMK * 2);
  u16* Vb  = (u16*)alloc((size_t)BN * DIMK * 2);
  u16* Vt  = (u16*)alloc((size_t)BN * DIMK * 2);
  float* kmfb    = (float*)alloc((size_t)BN * 4);
  float* partial = (float*)alloc((size_t)BB * DCHUNKS * 4);
  float* stats   = (float*)alloc((size_t)(BB + 1) * 4);

  const int nx = BN * DIMK;       // 4,194,304
  const int nw = DIMK * DIMK;     // 1,048,576

  // stats pipeline
  kmf_kernel<<<(BN + 255) / 256, 256, 0, stream>>>(mask, kmfb, BN);
  dist_partial_kernel<<<dim3(DCHUNKS, BB), 256, 0, stream>>>(dist, kmfb, partial);
  finalize_kernel<<<1, 256, 0, stream>>>(partial, kmfb, galpha, stats);

  // precision conversion
  f32_to_bf16_kernel<<<(nx + 255) / 256, 256, 0, stream>>>(x, xb, nx);
  wtrans_kernel<<<(nw + 255) / 256, 256, 0, stream>>>(wq, wqT, DIMK, DIMK);
  wtrans_kernel<<<(nw + 255) / 256, 256, 0, stream>>>(wk, wkT, DIMK, DIMK);
  wtrans_kernel<<<(nw + 255) / 256, 256, 0, stream>>>(wv, wvT, DIMK, DIMK);
  wtrans_kernel<<<(nw + 255) / 256, 256, 0, stream>>>(wo, woT, DIMK, DIMK);

  // projections (WMMA bf16)
  dim3 ggrid(BN / 16, DIMK / 64);
  gemm_bf16_kernel<true><<<ggrid, 32, 0, stream>>>(xb, wqT, Qb, BN, DIMK, DIMK);
  gemm_bf16_kernel<true><<<ggrid, 32, 0, stream>>>(xb, wkT, Kb, BN, DIMK, DIMK);
  gemm_bf16_kernel<true><<<ggrid, 32, 0, stream>>>(xb, wvT, Vb, BN, DIMK, DIMK);

  // K pair-swap/negate for the sin term; V transposed to d-major
  kswap_kernel<<<(nx / 2 + 255) / 256, 256, 0, stream>>>(Kb, Ksw, nx / 2);
  vtrans_kernel<<<(nx + 255) / 256, 256, 0, stream>>>(Vb, Vt, nx);

  // fused attention; writes bf16 attn-output over xb (x no longer needed)
  u16* attnb = xb;
  attn_kernel<<<dim3(NN / 16, HH, BB), 32, 0, stream>>>(
      Qb, Kb, Ksw, Vt, dist, kmfb, stats, head_omega, attnb);

  // output projection -> f32 d_out
  gemm_bf16_kernel<false><<<ggrid, 32, 0, stream>>>(attnb, woT, out, BN, DIMK, DIMK);
}
